// VisionAttention_42511586296288
// MI455X (gfx1250) — compile-verified, matrix operand order checked by reading, not run on hardware
//
#include <hip/hip_runtime.h>
#include <hip/hip_bf16.h>

// ---------------------------------------------------------------------------
// VisionAttention on gfx1250: all GEMM-like math via v_wmma_f32_16x16x32_f16.
// Software-pipelined GEMM (32x64 per wave), 64-key flash-attention tiles.
// ---------------------------------------------------------------------------

typedef __attribute__((ext_vector_type(16))) _Float16 v16h;
typedef __attribute__((ext_vector_type(8)))  float    v8f;

#define SEQ    4096
#define HIDDEN 1152
#define NHEADS 16
#define HDIM   72
#define DPAD   96      // head dim padded to 3 x 32 for the WMMA K loop
#define QKVN   3456

__device__ __forceinline__ v8f wmma32(v16h a, v16h b, v8f c) {
  // D = A(16x32 f16) x B(32x16 f16) + C(16x16 f32)
  return __builtin_amdgcn_wmma_f32_16x16x32_f16(false, a, false, b,
                                                (short)0, c, false, false);
}

// Per-lane fragment loader: elements 0..7 <- p[0..7], elements 8..15 <- p[16..23]
// (caller already folded in the +8*(lane>>4) half offset). Two b128 loads.
__device__ __forceinline__ v16h ldfrag(const _Float16* p) {
  v16h r;
  ((float4*)&r)[0] = *(const float4*)(p);
  ((float4*)&r)[1] = *(const float4*)(p + 16);
  return r;
}

// ---------------------------------------------------------------------------
__global__ void cvt_f16(const float* __restrict__ in, _Float16* __restrict__ out,
                        int n) {
  int i = blockIdx.x * blockDim.x + threadIdx.x;
  if (i < n) out[i] = (_Float16)in[i];
}

// LDS-tiled transpose+convert: WT[n*K + k] = (f16)W[k*N + n].
// Grid (K/32, N/32), 256 threads (32x8). Coalesced reads and writes.
__global__ __launch_bounds__(256) void transp_f16(
    const float* __restrict__ W, _Float16* __restrict__ WT, int K, int N) {
  __shared__ _Float16 tile[32][33];
  const int kb = blockIdx.x * 32, nb = blockIdx.y * 32;
  const int tx = threadIdx.x & 31, ty = threadIdx.x >> 5;
#pragma unroll
  for (int r = ty; r < 32; r += 8)
    tile[r][tx] = (_Float16)W[(size_t)(kb + r) * N + nb + tx];
  __syncthreads();
#pragma unroll
  for (int r = ty; r < 32; r += 8)
    WT[(size_t)(nb + r) * K + kb + tx] = tile[tx][r];
}

// ---------------------------------------------------------------------------
// C(MxN,f32) = A(MxK,f16 row-major) * BT(NxK,f16 row-major)^T + bias
// Block: 8 waves (4x2), each wave owns a 32x64 tile -> 128x128 block tile.
// K-loop is double-buffered in registers so loads overlap the 8 WMMAs.
struct GFrag { v16h a0, a1, b0, b1, b2, b3; };

__device__ __forceinline__ void gload(GFrag& f, const _Float16* ap0,
                                      const _Float16* ap1,
                                      const _Float16* const* bp, int kc) {
  f.a0 = ldfrag(ap0 + kc);
  f.a1 = ldfrag(ap1 + kc);
  f.b0 = ldfrag(bp[0] + kc);
  f.b1 = ldfrag(bp[1] + kc);
  f.b2 = ldfrag(bp[2] + kc);
  f.b3 = ldfrag(bp[3] + kc);
}

__device__ __forceinline__ void mma8(v8f acc[2][4], const GFrag& f) {
  acc[0][0] = wmma32(f.a0, f.b0, acc[0][0]);
  acc[0][1] = wmma32(f.a0, f.b1, acc[0][1]);
  acc[0][2] = wmma32(f.a0, f.b2, acc[0][2]);
  acc[0][3] = wmma32(f.a0, f.b3, acc[0][3]);
  acc[1][0] = wmma32(f.a1, f.b0, acc[1][0]);
  acc[1][1] = wmma32(f.a1, f.b1, acc[1][1]);
  acc[1][2] = wmma32(f.a1, f.b2, acc[1][2]);
  acc[1][3] = wmma32(f.a1, f.b3, acc[1][3]);
}

__global__ __launch_bounds__(256) void gemm_wmma(
    const _Float16* __restrict__ A, int lda,
    const _Float16* __restrict__ BT, int ldb,
    const float* __restrict__ bias,
    float* __restrict__ C, int ldc, int K) {
  const int lane = threadIdx.x & 31;
  const int w    = threadIdx.x >> 5;
  const int l15  = lane & 15;
  const int hf   = lane >> 4;
  const int wm   = w & 3;
  const int wn   = w >> 2;
  const int mb   = blockIdx.x * 128 + wm * 32;
  const int nb   = blockIdx.y * 128 + wn * 64;

  v8f acc[2][4] = {};
  const _Float16* ap0 = A + (size_t)(mb + l15) * lda + 8 * hf;
  const _Float16* ap1 = ap0 + (size_t)16 * lda;
  const _Float16* bp[4];
#pragma unroll
  for (int t = 0; t < 4; ++t)
    bp[t] = BT + (size_t)(nb + t * 16 + l15) * ldb + 8 * hf;

  GFrag cur, nxt;
  gload(cur, ap0, ap1, bp, 0);
#pragma unroll 2
  for (int kc = 32; kc < K; kc += 32) {
    gload(nxt, ap0, ap1, bp, kc);   // prefetch next chunk
    mma8(acc, cur);                 // overlaps with outstanding loads
    cur = nxt;
  }
  mma8(acc, cur);

#pragma unroll
  for (int i = 0; i < 2; ++i) {
#pragma unroll
    for (int t = 0; t < 4; ++t) {
      int col  = nb + t * 16 + l15;
      float bs = bias[col];
#pragma unroll
      for (int j = 0; j < 8; ++j)
        C[(size_t)(mb + i * 16 + j + 8 * hf) * ldc + col] = acc[i][t][j] + bs;
    }
  }
}

// ---------------------------------------------------------------------------
// RoPE for Q,K from qkv(f32) into head-major row-major f16, padded to DPAD.
// dp is the fastest index -> coalesced loads and stores.
__global__ void rope_qk(const float* __restrict__ qkv,
                        const float* __restrict__ cosb,
                        const float* __restrict__ sinb,
                        _Float16* __restrict__ Qp, _Float16* __restrict__ Kp) {
  int idx = blockIdx.x * blockDim.x + threadIdx.x;
  const int per = NHEADS * SEQ * DPAD;
  int which = idx / per;              // 0=Q, 1=K
  int r     = idx - which * per;
  int h     = r / (SEQ * DPAD);
  int r2    = r - h * (SEQ * DPAD);
  int s     = r2 / DPAD;
  int dp    = r2 - s * DPAD;

  float o = 0.f;
  if (dp < HDIM) {
    const float* base = qkv + (size_t)s * QKVN + which * HIDDEN + h * HDIM;
    float v   = base[dp];
    float rot = (dp < HDIM / 2) ? -base[dp + HDIM / 2] : base[dp - HDIM / 2];
    o = v * cosb[s * HDIM + dp] + rot * sinb[s * HDIM + dp];
  }
  (which ? Kp : Qp)[((size_t)h * SEQ + s) * DPAD + dp] = (_Float16)o;
}

// V: LDS-tiled transpose qkv[s][2*HIDDEN + h*72 + d] -> Vt[h][d][s], zero-pad
// d in [72,96). Grid (SEQ/32, DPAD/32, NHEADS), 256 threads (32x8).
__global__ __launch_bounds__(256) void v_transpose(const float* __restrict__ qkv,
                                                   _Float16* __restrict__ Vt) {
  __shared__ _Float16 tile[32][33];
  const int sb = blockIdx.x * 32, db = blockIdx.y * 32, h = blockIdx.z;
  const int tx = threadIdx.x & 31, ty = threadIdx.x >> 5;
#pragma unroll
  for (int r = ty; r < 32; r += 8) {
    int d   = db + tx;
    float v = (d < HDIM)
                ? qkv[(size_t)(sb + r) * QKVN + 2 * HIDDEN + h * HDIM + d]
                : 0.f;
    tile[r][tx] = (_Float16)v;
  }
  __syncthreads();
#pragma unroll
  for (int r = ty; r < 32; r += 8)
    Vt[((size_t)h * DPAD + db + r) * SEQ + sb + tx] = tile[tx][r];
}

// ---------------------------------------------------------------------------
// Flash attention: grid (SEQ/128, NHEADS), 8 waves/block, each wave = 16 rows,
// 64 keys per iteration. Online softmax; P repacked via per-wave LDS buffer.
__global__ __launch_bounds__(256) void attn_wmma(
    const _Float16* __restrict__ Qp, const _Float16* __restrict__ Kp,
    const _Float16* __restrict__ Vt, _Float16* __restrict__ O) {
  __shared__ _Float16 Plds[8][16 * 64];
  const int lane = threadIdx.x & 31;
  const int w    = threadIdx.x >> 5;
  const int l15  = lane & 15;
  const int hf   = lane >> 4;
  const int h    = blockIdx.y;
  const int q0   = (blockIdx.x * 8 + w) * 16;
  const float scale = 0.1178511302f;  // 72^-0.5

  v16h qa[3];
  {
    const _Float16* qrow = Qp + ((size_t)h * SEQ + q0 + l15) * DPAD + 8 * hf;
#pragma unroll
    for (int c = 0; c < 3; ++c) qa[c] = ldfrag(qrow + c * 32);
  }

  v8f acc[5] = {};
  float m[8], l[8];
#pragma unroll
  for (int j = 0; j < 8; ++j) { m[j] = -1e30f; l[j] = 0.f; }

  const _Float16* Kbase = Kp + ((size_t)h * SEQ + l15) * DPAD + 8 * hf;
  const _Float16* Vbase = Vt + ((size_t)h * DPAD + l15) * SEQ + 8 * hf;

  for (int kb = 0; kb < SEQ; kb += 64) {
    __builtin_prefetch(Kbase + (size_t)(kb + 64) * DPAD, 0, 1);
    // ---- S = Q K^T for 64 keys (four 16-wide column tiles), pipelined ----
    v8f s[4] = {};
    v16h kf[4];
#pragma unroll
    for (int t = 0; t < 4; ++t)
      kf[t] = ldfrag(Kbase + (size_t)(kb + t * 16) * DPAD);
#pragma unroll
    for (int c = 0; c < 3; ++c) {
      v16h kn[4];
      if (c < 2) {
#pragma unroll
        for (int t = 0; t < 4; ++t)
          kn[t] = ldfrag(Kbase + (size_t)(kb + t * 16) * DPAD + (c + 1) * 32);
      }
#pragma unroll
      for (int t = 0; t < 4; ++t) s[t] = wmma32(qa[c], kf[t], s[t]);
      if (c < 2) {
#pragma unroll
        for (int t = 0; t < 4; ++t) kf[t] = kn[t];
      }
    }

    // ---- online softmax (row reductions across 16 lanes of each half) ----
    float p[4][8];
#pragma unroll
    for (int j = 0; j < 8; ++j) {
      float v0 = s[0][j] * scale, v1 = s[1][j] * scale;
      float v2 = s[2][j] * scale, v3 = s[3][j] * scale;
      float mx = fmaxf(fmaxf(v0, v1), fmaxf(v2, v3));
      mx = fmaxf(mx, __shfl_xor(mx, 1));
      mx = fmaxf(mx, __shfl_xor(mx, 2));
      mx = fmaxf(mx, __shfl_xor(mx, 4));
      mx = fmaxf(mx, __shfl_xor(mx, 8));
      float mn    = fmaxf(m[j], mx);
      float alpha = __expf(m[j] - mn);
      p[0][j] = __expf(v0 - mn);
      p[1][j] = __expf(v1 - mn);
      p[2][j] = __expf(v2 - mn);
      p[3][j] = __expf(v3 - mn);
      float rs = (p[0][j] + p[1][j]) + (p[2][j] + p[3][j]);
      rs += __shfl_xor(rs, 1);
      rs += __shfl_xor(rs, 2);
      rs += __shfl_xor(rs, 4);
      rs += __shfl_xor(rs, 8);
      l[j] = l[j] * alpha + rs;
      m[j] = mn;
#pragma unroll
      for (int nt = 0; nt < 5; ++nt) acc[nt][j] *= alpha;
    }

    // ---- repack P (C-layout) -> A-fragment layout through LDS ----
#pragma unroll
    for (int j = 0; j < 8; ++j) {
      int M = j + 8 * hf;
#pragma unroll
      for (int t = 0; t < 4; ++t)
        Plds[w][M * 64 + t * 16 + l15] = (_Float16)p[t][j];
    }
    v16h pa[2];
#pragma unroll
    for (int kk = 0; kk < 2; ++kk) {
      ((float4*)&pa[kk])[0] =
          *(const float4*)(&Plds[w][l15 * 64 + kk * 32 + 8 * hf]);
      ((float4*)&pa[kk])[1] =
          *(const float4*)(&Plds[w][l15 * 64 + kk * 32 + 16 + 8 * hf]);
    }

    // ---- O += P * V  (5 d-tiles cover d = 0..79; [72,80) is zero-padded) ----
#pragma unroll
    for (int nt = 0; nt < 5; ++nt) {
      const _Float16* vr = Vbase + (size_t)(nt * 16) * SEQ + kb;
      v16h v0 = ldfrag(vr);
      v16h v1 = ldfrag(vr + 32);
      acc[nt] = wmma32(pa[0], v0, acc[nt]);
      acc[nt] = wmma32(pa[1], v1, acc[nt]);
    }
  }

  // ---- normalize and write f16 attention output (SEQ x HIDDEN row-major) ----
#pragma unroll
  for (int nt = 0; nt < 5; ++nt) {
    int d = nt * 16 + l15;
    if (d < HDIM) {
#pragma unroll
      for (int j = 0; j < 8; ++j) {
        int row = q0 + j + 8 * hf;
        O[(size_t)row * HIDDEN + h * HDIM + d] = (_Float16)(acc[nt][j] / l[j]);
      }
    }
  }
}

// ---------------------------------------------------------------------------
extern "C" void kernel_launch(void* const* d_in, const int* in_sizes, int n_in,
                              void* d_out, int out_size, void* d_ws, size_t ws_size,
                              hipStream_t stream) {
  (void)in_sizes; (void)n_in; (void)out_size; (void)ws_size;
  const float* hidden = (const float*)d_in[0];
  // d_in[1] = cu_seqlens: unused by the reference computation
  const float* cosb   = (const float*)d_in[2];
  const float* sinb   = (const float*)d_in[3];
  const float* Wqkv   = (const float*)d_in[4];
  const float* bqkv   = (const float*)d_in[5];
  const float* Wproj  = (const float*)d_in[6];
  const float* bproj  = (const float*)d_in[7];
  float* out = (float*)d_out;

  char* p = (char*)d_ws;
  _Float16* hid16  = (_Float16*)p; p += (size_t)SEQ * HIDDEN * 2;      // 9.0 MiB
  _Float16* WqkvT  = (_Float16*)p; p += (size_t)QKVN * HIDDEN * 2;     // 7.6 MiB
  _Float16* WprojT = (_Float16*)p; p += (size_t)HIDDEN * HIDDEN * 2;   // 2.5 MiB
  float*    qkvf   = (float*)p;    p += (size_t)SEQ * QKVN * 4;        // 54  MiB
  _Float16* Qp     = (_Float16*)p; p += (size_t)NHEADS * SEQ * DPAD * 2;
  _Float16* Kp     = (_Float16*)p; p += (size_t)NHEADS * SEQ * DPAD * 2;
  _Float16* Vt     = (_Float16*)p; p += (size_t)NHEADS * SEQ * DPAD * 2;
  _Float16* AO     = (_Float16*)p; p += (size_t)SEQ * HIDDEN * 2;

  cvt_f16<<<(SEQ * HIDDEN) / 256, 256, 0, stream>>>(hidden, hid16, SEQ * HIDDEN);
  transp_f16<<<dim3(HIDDEN / 32, QKVN / 32), 256, 0, stream>>>(
      Wqkv, WqkvT, HIDDEN, QKVN);
  transp_f16<<<dim3(HIDDEN / 32, HIDDEN / 32), 256, 0, stream>>>(
      Wproj, WprojT, HIDDEN, HIDDEN);

  // qkv = hidden @ Wqkv + bqkv   (f32 out for accurate RoPE)
  gemm_wmma<<<dim3(SEQ / 128, QKVN / 128), 256, 0, stream>>>(
      hid16, HIDDEN, WqkvT, HIDDEN, bqkv, qkvf, QKVN, HIDDEN);

  rope_qk<<<(2 * NHEADS * SEQ * DPAD) / 256, 256, 0, stream>>>(
      qkvf, cosb, sinb, Qp, Kp);
  v_transpose<<<dim3(SEQ / 32, DPAD / 32, NHEADS), 256, 0, stream>>>(qkvf, Vt);

  attn_wmma<<<dim3(SEQ / 128, NHEADS), 256, 0, stream>>>(Qp, Kp, Vt, AO);

  // out = attn @ Wproj + bproj
  gemm_wmma<<<dim3(SEQ / 128, HIDDEN / 128), 256, 0, stream>>>(
      AO, HIDDEN, WprojT, HIDDEN, bproj, out, HIDDEN, HIDDEN);
}
